// ImpSentenceModel_20023137534912
// MI455X (gfx1250) — compile-verified, matrix operand order checked by reading, not run on hardware
//
#include <hip/hip_runtime.h>
#include <hip/hip_bf16.h>

typedef __attribute__((ext_vector_type(16))) _Float16 v16h;
typedef __attribute__((ext_vector_type(8)))  _Float16 v8h;
typedef __attribute__((ext_vector_type(8)))  float    v8f;
typedef __attribute__((ext_vector_type(4)))  unsigned u32x4;
typedef __attribute__((ext_vector_type(4)))  int      i32x4;
typedef __attribute__((ext_vector_type(8)))  int      i32x8;

union U16 { v16h v; v8h h[2]; };

constexpr int Bc  = 32;     // batch
constexpr int Tc  = 4096;   // tokens
constexpr int Ec  = 256;    // embedding dim
constexpr int Hc  = 256;    // hidden dim
constexpr int Sc  = 128;    // max sentences
constexpr int G4H = 1024;   // 4*H

// workspace byte offsets
constexpr size_t LE16_OFF = 0;                    // [S*B, E] f16          (2 MB)
constexpr size_t WI16_OFF = 2097152;              // [4H, E]  f16          (512 KB)
constexpr size_t WH16_OFF = 2621440;              // [4H, H]  f16          (512 KB)
constexpr size_t BIAS_OFF = 3145728;              // [4H]     f32          (4 KB)
constexpr size_t XP_OFF   = 3149824;              // fragment-major x_proj (16 MB)
constexpr size_t H16_OFF  = 19927040;             // 2x[B, H] f16          (32 KB)
constexpr size_t CTR_OFF  = 19959808;             // barrier counter

__device__ __forceinline__ float sigmoidf_(float x) {
  return 1.0f / (1.0f + __expf(-x));
}

// ---------------- k0: weight converts + bias sum + state init ----------------
__global__ void k0_prep(const float* __restrict__ Wi, const float* __restrict__ Wh,
                        const float* __restrict__ bi, const float* __restrict__ bh,
                        _Float16* __restrict__ wi16, _Float16* __restrict__ wh16,
                        float* __restrict__ bias, _Float16* __restrict__ h16,
                        unsigned* __restrict__ ctr) {
  int i = blockIdx.x * blockDim.x + threadIdx.x;
  if (i < G4H * Ec)    wi16[i] = (_Float16)Wi[i];
  if (i < G4H * Hc)    wh16[i] = (_Float16)Wh[i];
  if (i < G4H)         bias[i] = bi[i] + bh[i];
  if (i < 2 * Bc * Hc) h16[i] = (_Float16)0.0f;
  if (i == 0)          *ctr = 0u;
}

// ---------------- k1: embedding gather + segment sum -> [S,B,E] f16 ----------
__global__ void k1_segsum(const int* __restrict__ para, const int* __restrict__ slen,
                          const float* __restrict__ emb, _Float16* __restrict__ le16) {
  int bs = blockIdx.x;
  int b = bs / Sc, s = bs % Sc;
  int e = threadIdx.x;
  int start = 0;
  for (int i = 0; i < s; ++i) start += slen[b * Sc + i];
  int len = slen[b * Sc + s];
  int st = start < Tc ? start : Tc;
  int en = start + len; en = en < Tc ? en : Tc;
  const int* prow = para + (size_t)b * Tc;
  float acc = 0.0f;
  if (st < en) {
    int tok = prow[st];
    for (int t = st; t < en; ++t) {
      int nxt = (t + 1 < en) ? prow[t + 1] : tok;
      __builtin_prefetch(&emb[(size_t)nxt * Ec + e], 0, 0);  // global_prefetch_b8
      acc += emb[(size_t)tok * Ec + e];
      tok = nxt;
    }
  }
  le16[((size_t)s * Bc + b) * Ec + e] = (_Float16)acc;
}

// ---------------- k2: x_proj = xs @ Wi^T + (bi+bh), WMMA f16 -----------------
// M = S*B = 4096, N = 4H = 1024, K = E = 256. One 16x16 tile per wave.
// Output stored FRAGMENT-MAJOR: xp[((tm*64+tn)*32 + lane)*8 + r]
__global__ void k2_xproj(const _Float16* __restrict__ A, const _Float16* __restrict__ Wih,
                         const float* __restrict__ bias, float* __restrict__ xp) {
  int lane = threadIdx.x & 31;
  int wave = threadIdx.x >> 5;
  int tile = blockIdx.x * (blockDim.x >> 5) + wave;
  int tm = tile >> 6;        // 0..255
  int tn = tile & 63;        // 0..63
  int g2  = lane >> 4;
  int l15 = lane & 15;
  const _Float16* ar = A   + (size_t)(tm * 16 + l15) * Ec;
  const _Float16* br = Wih + (size_t)(tn * 16 + l15) * Ec;
  v8f acc = {};
  for (int kb = 0; kb < Ec; kb += 32) {
    U16 ua, ub;
    ua.h[0] = *(const v8h*)(ar + kb + 8 * g2);
    ua.h[1] = *(const v8h*)(ar + kb + 8 * g2 + 16);
    ub.h[0] = *(const v8h*)(br + kb + 8 * g2);
    ub.h[1] = *(const v8h*)(br + kb + 8 * g2 + 16);
    acc = __builtin_amdgcn_wmma_f32_16x16x32_f16(false, ua.v, false, ub.v,
                                                 (short)0, acc, false, false);
  }
  acc = acc + bias[tn * 16 + l15];                 // n depends only on lane
  *(v8f*)(xp + ((size_t)tile * 32 + lane) * 8) = acc;
}

// ---------------- TDM: stage one 16-row x 256-half 2D tile into LDS ----------
__device__ __forceinline__ void tdm_load_tile(unsigned lds_byte_addr,
                                              unsigned long long gaddr) {
  u32x4 g0;
  g0[0] = 1u;                                      // count=1, is_restore=0
  g0[1] = lds_byte_addr;                           // lds_addr
  g0[2] = (unsigned)(gaddr & 0xFFFFFFFFull);       // global_addr[31:0]
  g0[3] = (unsigned)((gaddr >> 32) & 0x1FFFFFFull) // global_addr[56:32]
          | (2u << 30);                            // type=2 ("image")
  i32x8 g1;
  g1[0] = (1 << 16);                               // data_size=1 (2B), mask=0
  g1[1] = (256 << 16);                             // tensor_dim0[15:0]=256 @63:48
  g1[2] = (16 << 16);                              // tensor_dim1[15:0]=16  @95:80
  g1[3] = (256 << 16);                             // tile_dim0=256        @127:112
  g1[4] = 16;                                      // tile_dim1=16, tile_dim2=0
  g1[5] = 256;                                     // tensor_dim0_stride=256
  g1[6] = 0;
  g1[7] = 0;
  i32x4 z4 = {};
#if __has_include(<hip/amd_detail/amd_gfx1250_TDM.h>)
  i32x8 z8 = {};
  __builtin_amdgcn_tensor_load_to_lds(g0, g1, z4, z4, z8, 0);
#else
  __builtin_amdgcn_tensor_load_to_lds(g0, g1, z4, z4, 0);
#endif
}

// ---------------- k3: persistent WMMA LSTM recurrence ------------------------
// 16 WGs x 64 threads. WG owns 16 hidden units (4 gate row-blocks of Wh -> LDS
// via Tensor Data Mover). Double-buffered global h (f16); grid barrier per step.
__global__ void __launch_bounds__(64) k3_lstm(const float* __restrict__ xp,
                                              const _Float16* __restrict__ wh16,
                                              const int* __restrict__ plen,
                                              _Float16* __restrict__ h16,
                                              unsigned* __restrict__ ctr,
                                              float* __restrict__ out) {
  __shared__ _Float16 whs[64 * Hc];   // 32 KB: rows = gate*16 + j_local
  int j0   = blockIdx.x * 16;
  int lane = threadIdx.x & 31;
  int w    = threadIdx.x >> 5;        // m-tile (batch 0-15 / 16-31)
  int g2   = lane >> 4;
  int jl   = lane & 15;

  // stage this WG's four 16x256 f16 gate slices of Wh into LDS with the TDM
  if (w == 0) {
    unsigned lds0 = (unsigned)(unsigned long long)(uintptr_t)&whs[0];
#pragma unroll
    for (int gate = 0; gate < 4; ++gate) {
      unsigned long long ga = (unsigned long long)(uintptr_t)
          (wh16 + (size_t)(gate * Hc + j0) * Hc);
      tdm_load_tile(lds0 + (unsigned)(gate * 16 * Hc * 2), ga);
    }
    __builtin_amdgcn_s_wait_tensorcnt(0);
  }
  __syncthreads();

  float cst[8] = {};
  int pl[8];
#pragma unroll
  for (int r = 0; r < 8; ++r) pl[r] = plen[w * 16 + r + 8 * g2];

  for (int s = 0; s < Sc; ++s) {
    const _Float16* hcur = h16 + (size_t)(s & 1) * (Bc * Hc);
    _Float16*       hnxt = h16 + (size_t)((s + 1) & 1) * (Bc * Hc);

    // accumulators seeded from fragment-major x_proj (biases already folded in)
    v8f acc[4];
    {
      int tm3 = s * 2 + w;
#pragma unroll
      for (int g = 0; g < 4; ++g) {
        int tile = tm3 * 64 + (g * 16 + blockIdx.x);
        acc[g] = *(const v8f*)(xp + ((size_t)tile * 32 + lane) * 8);
      }
    }

    const _Float16* ar = hcur + (size_t)(w * 16 + (lane & 15)) * Hc;
    for (int kb = 0; kb < Hc; kb += 32) {
      U16 ua;
      ua.h[0] = *(const v8h*)(ar + kb + 8 * g2);
      ua.h[1] = *(const v8h*)(ar + kb + 8 * g2 + 16);
#pragma unroll
      for (int g = 0; g < 4; ++g) {
        U16 ub;
        const _Float16* br = &whs[(size_t)(g * 16 + (lane & 15)) * Hc];
        ub.h[0] = *(const v8h*)(br + kb + 8 * g2);
        ub.h[1] = *(const v8h*)(br + kb + 8 * g2 + 16);
        acc[g] = __builtin_amdgcn_wmma_f32_16x16x32_f16(false, ua.v, false, ub.v,
                                                        (short)0, acc[g], false, false);
      }
    }

    // LSTM cell (fp32), masked update, write h_next + output slice
#pragma unroll
    for (int r = 0; r < 8; ++r) {
      int b = w * 16 + r + 8 * g2;
      float ig = sigmoidf_(acc[0][r]);
      float fg = sigmoidf_(acc[1][r]);
      float gg = tanhf(acc[2][r]);
      float og = sigmoidf_(acc[3][r]);
      float cn = fg * cst[r] + ig * gg;
      float hn = og * tanhf(cn);
      size_t ho = (size_t)b * Hc + (j0 + jl);
      size_t oo = ((size_t)s * Bc + b) * Hc + (j0 + jl);
      if (s < pl[r]) {
        cst[r] = cn;
        hnxt[ho] = (_Float16)hn;
        out[oo] = hn;
      } else {
        hnxt[ho] = hcur[ho];
        out[oo] = 0.0f;
      }
    }

    // grid-wide barrier (16 WGs, monotonic counter: no reset needed)
    __syncthreads();
    if (threadIdx.x == 0) {
      __hip_atomic_fetch_add(ctr, 1u, __ATOMIC_ACQ_REL, __HIP_MEMORY_SCOPE_AGENT);
      unsigned tgt = 16u * (unsigned)(s + 1);
      while (__hip_atomic_load(ctr, __ATOMIC_ACQUIRE, __HIP_MEMORY_SCOPE_AGENT) < tgt) {
        __builtin_amdgcn_s_sleep(2);
      }
    }
    __syncthreads();
  }
}

extern "C" void kernel_launch(void* const* d_in, const int* in_sizes, int n_in,
                              void* d_out, int out_size, void* d_ws, size_t ws_size,
                              hipStream_t stream) {
  (void)in_sizes; (void)n_in; (void)out_size; (void)ws_size;
  const int*   para = (const int*)d_in[0];
  const int*   slen = (const int*)d_in[1];
  const int*   plen = (const int*)d_in[2];
  const float* emb  = (const float*)d_in[3];
  const float* Wi   = (const float*)d_in[4];
  const float* Wh   = (const float*)d_in[5];
  const float* bi   = (const float*)d_in[6];
  const float* bh   = (const float*)d_in[7];
  float* out = (float*)d_out;

  char* ws = (char*)d_ws;
  _Float16* le16 = (_Float16*)(ws + LE16_OFF);
  _Float16* wi16 = (_Float16*)(ws + WI16_OFF);
  _Float16* wh16 = (_Float16*)(ws + WH16_OFF);
  float*    bias = (float*)(ws + BIAS_OFF);
  float*    xpb  = (float*)(ws + XP_OFF);
  _Float16* h16  = (_Float16*)(ws + H16_OFF);
  unsigned* ctr  = (unsigned*)(ws + CTR_OFF);

  k0_prep<<<(G4H * Ec) / 256, 256, 0, stream>>>(Wi, Wh, bi, bh, wi16, wh16, bias, h16, ctr);
  k1_segsum<<<Bc * Sc, 256, 0, stream>>>(para, slen, emb, le16);
  // tiles: (4096/16) * (1024/16) = 16384; 8 waves per 256-thread block
  k2_xproj<<<16384 / 8, 256, 0, stream>>>(le16, wi16, bias, xpb);
  k3_lstm<<<16, 64, 0, stream>>>(xpb, wh16, plen, h16, ctr, out);
}
